// GPT2Block_62818191671337
// MI455X (gfx1250) — compile-verified, matrix operand order checked by reading, not run on hardware
//
#include <hip/hip_runtime.h>
#include <hip/hip_bf16.h>

typedef __bf16 bf16;
typedef __attribute__((ext_vector_type(16))) __bf16 v16bf;
typedef __attribute__((ext_vector_type(8)))  __bf16 v8bf;
typedef __attribute__((ext_vector_type(8)))  float  v8f;
typedef int v4i __attribute__((vector_size(16)));

#define D_EMBD 1024
#define N_HEAD 16
#define HEAD_DIM 64

#if __has_builtin(__builtin_amdgcn_global_load_async_to_lds_b128)
#define HAS_ASYNC_LDS 1
#else
#define HAS_ASYNC_LDS 0
#endif

// 16-byte global -> LDS copy; async (ASYNCcnt-tracked) when available.
__device__ __forceinline__ void cp_lds16(void* lds, const void* g) {
#if HAS_ASYNC_LDS
  __builtin_amdgcn_global_load_async_to_lds_b128(
      (__attribute__((address_space(1))) v4i*)(g),
      (__attribute__((address_space(3))) v4i*)(lds), 0, 0);
#else
  *(v8bf*)lds = *(const v8bf*)g;
#endif
}

__device__ __forceinline__ void wait_async0() {
#if HAS_ASYNC_LDS
#if __has_builtin(__builtin_amdgcn_s_wait_asynccnt)
  __builtin_amdgcn_s_wait_asynccnt(0);
#else
  asm volatile("s_wait_asynccnt 0x0" ::: "memory");
#endif
#endif
}

// ---------------------------------------------------------------------------
// Fragment load: 16x32 bf16 tile, row-major with leading dim `ld`.
// Per CDNA5 ISA 16-bit A-matrix layout: lane r (0..15) holds row r;
// VGPRs 0-3 hold K = h*8 .. h*8+7, VGPRs 4-7 hold K = 16+h*8 .. 16+h*8+7
// where h = lane>>4.  B fragments (from W^T stored [N,K]) use the same
// pattern with lane -> column N.
// ---------------------------------------------------------------------------
__device__ __forceinline__ v16bf frag_ld(const bf16* tile, int ld) {
  const int lane = threadIdx.x & 31;
  const int r = lane & 15;
  const int h = lane >> 4;
  const bf16* p = tile + (size_t)r * ld + h * 8;
  v8bf lo = *(const v8bf*)(p);
  v8bf hi = *(const v8bf*)(p + 16);
  v16bf o;
#pragma unroll
  for (int i = 0; i < 8; ++i) { o[i] = lo[i]; o[8 + i] = hi[i]; }
  return o;
}

__device__ __forceinline__ v8f wmma_bf16(v16bf a, v16bf b, v8f c) {
  return __builtin_amdgcn_wmma_f32_16x16x32_bf16(
      /*neg_a=*/false, a, /*neg_b=*/false, b,
      /*c_mod=*/(short)0, c, /*reuse_a=*/false, /*reuse_b=*/false);
}

// ---------------------------------------------------------------------------
// Weight transpose + fp32->bf16 convert:  W[K,N] fp32  ->  Wt[N,K] bf16
// ---------------------------------------------------------------------------
__global__ __launch_bounds__(256) void transpose_conv(
    const float* __restrict__ W, bf16* __restrict__ Wt, int K, int N) {
  __shared__ float tile[32][33];
  const int n0 = blockIdx.x * 32;
  const int k0 = blockIdx.y * 32;
  for (int i = threadIdx.y; i < 32; i += 8)
    tile[i][threadIdx.x] = W[(size_t)(k0 + i) * N + n0 + threadIdx.x];
  __syncthreads();
  for (int i = threadIdx.y; i < 32; i += 8)
    Wt[(size_t)(n0 + i) * K + k0 + threadIdx.x] = (bf16)tile[threadIdx.x][i];
}

// ---------------------------------------------------------------------------
// LayerNorm over D=1024, one block (256 threads) per row, bf16 output.
// ---------------------------------------------------------------------------
__global__ __launch_bounds__(256) void ln_kernel(
    const float* __restrict__ x, const float* __restrict__ g,
    const float* __restrict__ bb, bf16* __restrict__ out) {
  __shared__ float red[256];
  const int t = threadIdx.x;
  const float* row = x + (size_t)blockIdx.x * D_EMBD;
  float v[4];
  float s = 0.f;
#pragma unroll
  for (int i = 0; i < 4; ++i) { v[i] = row[t + i * 256]; s += v[i]; }
  red[t] = s;
  __syncthreads();
  for (int o = 128; o > 0; o >>= 1) {
    if (t < o) red[t] += red[t + o];
    __syncthreads();
  }
  const float mu = red[0] * (1.0f / D_EMBD);
  __syncthreads();
  float s2 = 0.f;
#pragma unroll
  for (int i = 0; i < 4; ++i) { float d = v[i] - mu; s2 += d * d; }
  red[t] = s2;
  __syncthreads();
  for (int o = 128; o > 0; o >>= 1) {
    if (t < o) red[t] += red[t + o];
    __syncthreads();
  }
  const float rstd = rsqrtf(red[0] * (1.0f / D_EMBD) + 1e-5f);
  bf16* orow = out + (size_t)blockIdx.x * D_EMBD;
#pragma unroll
  for (int i = 0; i < 4; ++i) {
    const int c = t + i * 256;
    orow[c] = (bf16)((v[i] - mu) * rstd * g[c] + bb[c]);
  }
}

// ---------------------------------------------------------------------------
// WMMA GEMM:  C[M,N] = A[M,K] (bf16, row-major) @ Wt[N,K]^T (bf16) + bias
// 256 threads = 8 waves; block tile 128x128; wave tile 32x64 (2x4 frags).
// K-tiles (128x32 of A and of Wt) are staged into LDS with async
// global->LDS copies (ASYNCcnt), double-buffered against the WMMA pipe.
// EPI: 0 = bias -> bf16; 1 = bias + fp32 residual -> fp32; 2 = bias+GELU -> bf16
// ---------------------------------------------------------------------------
template <int EPI>
__global__ __launch_bounds__(256) void gemm_bf16_kernel(
    const bf16* __restrict__ A, const bf16* __restrict__ Bt,
    const float* __restrict__ bias, const float* __restrict__ resid,
    float* __restrict__ Cf, bf16* __restrict__ Cb, int M, int N, int K) {
  __shared__ __align__(16) bf16 As[2][128][32];
  __shared__ __align__(16) bf16 Bs[2][128][32];

  const int w = threadIdx.x >> 5;
  const int wr = w >> 1;           // 0..3 -> M rows within block
  const int wc = w & 1;            // 0..1 -> N cols within block
  const int m0 = blockIdx.y * 128 + wr * 32;
  const int n0 = blockIdx.x * 128 + wc * 64;

  // Cooperative tile prefetch: 256 threads x (2 A-chunks + 2 B-chunks of 16B)
  const int prow = threadIdx.x >> 1;        // 0..127
  const int pcol = (threadIdx.x & 1) * 16;  // 0 or 16
  const bf16* gA = A + (size_t)(blockIdx.y * 128 + prow) * K + pcol;
  const bf16* gB = Bt + (size_t)(blockIdx.x * 128 + prow) * K + pcol;

  auto prefetch = [&](int buf, int kb) {
    cp_lds16(&As[buf][prow][pcol],     gA + kb);
    cp_lds16(&As[buf][prow][pcol + 8], gA + kb + 8);
    cp_lds16(&Bs[buf][prow][pcol],     gB + kb);
    cp_lds16(&Bs[buf][prow][pcol + 8], gB + kb + 8);
  };

  v8f acc[2][4];
#pragma unroll
  for (int i = 0; i < 2; ++i)
#pragma unroll
    for (int j = 0; j < 4; ++j) acc[i][j] = v8f{0.f, 0.f, 0.f, 0.f, 0.f, 0.f, 0.f, 0.f};

  prefetch(0, 0);
  int buf = 0;
  for (int kb = 0; kb < K; kb += 32) {
    wait_async0();        // our tile-copy group for `buf` has landed in LDS
    __syncthreads();      // everyone's copies landed; nobody still reads buf^1
    if (kb + 32 < K) prefetch(buf ^ 1, kb + 32);  // stream next tile

    v16bf a0 = frag_ld(&As[buf][wr * 32][0], 32);
    v16bf a1 = frag_ld(&As[buf][wr * 32 + 16][0], 32);
#pragma unroll
    for (int j = 0; j < 4; ++j) {
      v16bf bj = frag_ld(&Bs[buf][wc * 64 + j * 16][0], 32);
      acc[0][j] = wmma_bf16(a0, bj, acc[0][j]);
      acc[1][j] = wmma_bf16(a1, bj, acc[1][j]);
    }
    buf ^= 1;
  }

  const int lane = threadIdx.x & 31;
  const int r = lane & 15;
  const int h8 = (lane >> 4) * 8;
#pragma unroll
  for (int i = 0; i < 2; ++i)
#pragma unroll
    for (int j = 0; j < 4; ++j)
#pragma unroll
      for (int jj = 0; jj < 8; ++jj) {
        const int m = m0 + i * 16 + h8 + jj;
        const int n = n0 + j * 16 + r;
        float v = acc[i][j][jj] + bias[n];
        if constexpr (EPI == 1) {
          v += resid[(size_t)m * N + n];
          Cf[(size_t)m * N + n] = v;
        } else if constexpr (EPI == 2) {
          v = 0.5f * v * (1.0f + erff(v * 0.70710678118654752f));
          Cb[(size_t)m * N + n] = (bf16)v;
        } else {
          Cb[(size_t)m * N + n] = (bf16)v;
        }
      }
}

// ---------------------------------------------------------------------------
// Flash-style causal attention.  qkv: bf16 [B,S,3*D]; out: bf16 [B,S,D].
// Grid: (S/64, N_HEAD, B); 128 threads = 4 waves; wave w owns 16 query rows.
// ---------------------------------------------------------------------------
__global__ __launch_bounds__(128) void attn_kernel(
    const bf16* __restrict__ qkv, bf16* __restrict__ out, int S) {
  __shared__ __align__(16) bf16 VT[64][72];      // V^T tile: [hd][key]
  __shared__ __align__(16) bf16 P[4][16][64];    // per-wave P tile

  const int b = blockIdx.z;
  const int hh = blockIdx.y;
  const int q0 = blockIdx.x * 64;
  const int w = threadIdx.x >> 5;
  const int lane = threadIdx.x & 31;
  const int r = lane & 15;
  const int h8 = (lane >> 4) * 8;

  const int ldq = 3 * D_EMBD;
  const bf16* base = qkv + (size_t)b * S * ldq + hh * HEAD_DIM;
  const bf16* qb = base;
  const bf16* kbp = base + D_EMBD;
  const bf16* vbp = base + 2 * D_EMBD;

  // Q fragments for this wave's 16 rows (hd = 64 -> 2 chunks of K=32)
  v16bf aq[2];
#pragma unroll
  for (int kk = 0; kk < 2; ++kk)
    aq[kk] = frag_ld(qb + (size_t)(q0 + w * 16) * ldq + kk * 32, ldq);

  v8f o[4];
  float mi[8], li[8];
#pragma unroll
  for (int nf = 0; nf < 4; ++nf) o[nf] = v8f{0.f, 0.f, 0.f, 0.f, 0.f, 0.f, 0.f, 0.f};
#pragma unroll
  for (int jj = 0; jj < 8; ++jj) { mi[jj] = -1e30f; li[jj] = 0.f; }

  const int kend = q0 + 64;  // exclusive, uniform across the block
  for (int kb0 = 0; kb0 < kend; kb0 += 64) {
    __syncthreads();  // protect VT / P reuse from previous iteration
    // Stage V^T into LDS: 64 keys x 64 hd; 2 threads per key row.
    {
      const int key = threadIdx.x >> 1;
      const int c0 = (threadIdx.x & 1) * 32;
      const bf16* vr = vbp + (size_t)(kb0 + key) * ldq + c0;
#pragma unroll
      for (int i = 0; i < 4; ++i) {
        v8bf ch = *(const v8bf*)(vr + i * 8);
#pragma unroll
        for (int e = 0; e < 8; ++e) VT[c0 + i * 8 + e][key] = ch[e];
      }
    }
    __syncthreads();

    // Scores: S = Q @ K^T  (K rows are contiguous over hd -> direct B frags)
    v8f sc[4];
#pragma unroll
    for (int j = 0; j < 4; ++j) sc[j] = v8f{0.f, 0.f, 0.f, 0.f, 0.f, 0.f, 0.f, 0.f};
#pragma unroll
    for (int j = 0; j < 4; ++j)
#pragma unroll
      for (int kk = 0; kk < 2; ++kk) {
        v16bf bk = frag_ld(kbp + (size_t)(kb0 + j * 16) * ldq + kk * 32, ldq);
        sc[j] = wmma_bf16(aq[kk], bk, sc[j]);
      }

    // Online softmax update (rows live one-per-VGPR across 16 lanes)
#pragma unroll
    for (int jj = 0; jj < 8; ++jj) {
      const int qrow = q0 + w * 16 + h8 + jj;
#pragma unroll
      for (int j = 0; j < 4; ++j) {
        const int key = kb0 + j * 16 + r;
        const float sv = sc[j][jj] * 0.125f;  // 1/sqrt(64)
        sc[j][jj] = (key <= qrow) ? sv : -1e30f;
      }
      float mx = fmaxf(fmaxf(sc[0][jj], sc[1][jj]), fmaxf(sc[2][jj], sc[3][jj]));
#pragma unroll
      for (int d = 1; d < 16; d <<= 1) mx = fmaxf(mx, __shfl_xor(mx, d, 32));
      const float nm = fmaxf(mi[jj], mx);
      const float alpha = __expf(mi[jj] - nm);
      float rs = 0.f;
#pragma unroll
      for (int j = 0; j < 4; ++j) {
        const float p = __expf(sc[j][jj] - nm);
        P[w][h8 + jj][j * 16 + r] = (bf16)p;
        rs += p;
      }
#pragma unroll
      for (int d = 1; d < 16; d <<= 1) rs += __shfl_xor(rs, d, 32);
      li[jj] = li[jj] * alpha + rs;
      mi[jj] = nm;
#pragma unroll
      for (int nf = 0; nf < 4; ++nf) o[nf][jj] *= alpha;
    }
    __syncthreads();  // P tile write -> A-fragment reload

    // O += P @ V  (A frags from LDS P; B frags from LDS V^T)
#pragma unroll
    for (int kk = 0; kk < 2; ++kk) {
      v16bf ap = frag_ld(&P[w][0][0] + kk * 32, 64);
#pragma unroll
      for (int nf = 0; nf < 4; ++nf) {
        v16bf bv = frag_ld(&VT[nf * 16][0] + kk * 32, 72);
        o[nf] = wmma_bf16(ap, bv, o[nf]);
      }
    }
  }

  // Normalize and write bf16 output
#pragma unroll
  for (int jj = 0; jj < 8; ++jj) {
    const float rl = 1.0f / li[jj];
    const size_t row = (size_t)b * S + q0 + w * 16 + h8 + jj;
#pragma unroll
    for (int nf = 0; nf < 4; ++nf)
      out[row * D_EMBD + hh * HEAD_DIM + nf * 16 + r] = (bf16)(o[nf][jj] * rl);
  }
}

// ---------------------------------------------------------------------------
extern "C" void kernel_launch(void* const* d_in, const int* in_sizes, int n_in,
                              void* d_out, int out_size, void* d_ws, size_t ws_size,
                              hipStream_t stream) {
  const float* x      = (const float*)d_in[0];
  const float* ln1_g  = (const float*)d_in[1];
  const float* ln1_b  = (const float*)d_in[2];
  const float* w_attn = (const float*)d_in[3];
  const float* b_attn = (const float*)d_in[4];
  const float* w_ap   = (const float*)d_in[5];
  const float* b_ap   = (const float*)d_in[6];
  const float* ln2_g  = (const float*)d_in[7];
  const float* ln2_b  = (const float*)d_in[8];
  const float* w_fc   = (const float*)d_in[9];
  const float* b_fc   = (const float*)d_in[10];
  const float* w_fp   = (const float*)d_in[11];
  const float* b_fp   = (const float*)d_in[12];
  float* out = (float*)d_out;

  const int D = D_EMBD;
  const int R = in_sizes[0] / D;  // B*S
  const int B = 2;
  const int S = R / B;

  char* ws = (char*)d_ws;
  auto alloc = [&](size_t bytes) {
    char* p = ws;
    ws += (bytes + 255) & ~(size_t)255;
    return (void*)p;
  };
  bf16*  h1      = (bf16*)alloc((size_t)R * D * 2);
  bf16*  Wt_attn = (bf16*)alloc((size_t)3 * D * D * 2);
  bf16*  Wt_ap   = (bf16*)alloc((size_t)D * D * 2);
  bf16*  Wt_fc   = (bf16*)alloc((size_t)4 * D * D * 2);
  bf16*  Wt_fp   = (bf16*)alloc((size_t)4 * D * D * 2);
  bf16*  qkv     = (bf16*)alloc((size_t)R * 3 * D * 2);
  bf16*  aout    = (bf16*)alloc((size_t)R * D * 2);
  float* x1      = (float*)alloc((size_t)R * D * 4);
  bf16*  h2      = (bf16*)alloc((size_t)R * D * 2);
  bf16*  mbuf    = (bf16*)alloc((size_t)R * 4 * D * 2);

  dim3 tb(32, 8);
  transpose_conv<<<dim3(3 * D / 32, D / 32), tb, 0, stream>>>(w_attn, Wt_attn, D, 3 * D);
  transpose_conv<<<dim3(D / 32, D / 32), tb, 0, stream>>>(w_ap, Wt_ap, D, D);
  transpose_conv<<<dim3(4 * D / 32, D / 32), tb, 0, stream>>>(w_fc, Wt_fc, D, 4 * D);
  transpose_conv<<<dim3(D / 32, 4 * D / 32), tb, 0, stream>>>(w_fp, Wt_fp, 4 * D, D);

  ln_kernel<<<R, 256, 0, stream>>>(x, ln1_g, ln1_b, h1);

  gemm_bf16_kernel<0><<<dim3(3 * D / 128, R / 128), 256, 0, stream>>>(
      h1, Wt_attn, b_attn, nullptr, nullptr, qkv, R, 3 * D, D);

  attn_kernel<<<dim3(S / 64, N_HEAD, B), 128, 0, stream>>>(qkv, aout, S);

  gemm_bf16_kernel<1><<<dim3(D / 128, R / 128), 256, 0, stream>>>(
      aout, Wt_ap, b_ap, x, x1, nullptr, R, D, D);

  ln_kernel<<<R, 256, 0, stream>>>(x1, ln2_g, ln2_b, h2);

  gemm_bf16_kernel<2><<<dim3(4 * D / 128, R / 128), 256, 0, stream>>>(
      h2, Wt_fc, b_fc, nullptr, nullptr, mbuf, R, 4 * D, D);

  gemm_bf16_kernel<1><<<dim3(D / 128, R / 128), 256, 0, stream>>>(
      mbuf, Wt_fp, b_fp, x1, out, nullptr, R, D, 4 * D);
}
